// GAT_22110491639989
// MI455X (gfx1250) — compile-verified, matrix operand order checked by reading, not run on hardware
//
#include <hip/hip_runtime.h>

#define NNODES 20000
#define E0     320000
#define ETOT   340000
#define NEG_SLOPE 0.2f
#define EPS_F  1e-16f
#define BSTR   40      // LDS B-tile row stride in halfs (80B, 16B-aligned chunks)

typedef __attribute__((ext_vector_type(16))) _Float16 v16h;
typedef __attribute__((ext_vector_type(8)))  _Float16 v8h;
typedef __attribute__((ext_vector_type(8)))  float    v8f;

// ---------------------------------------------------------------- utilities
__global__ void zero_f32(float* p, int n) {
    int i = blockIdx.x * blockDim.x + threadIdx.x;
    if (i < n) p[i] = 0.f;
}
__global__ void fill_u32(unsigned* p, unsigned v, int n) {
    int i = blockIdx.x * blockDim.x + threadIdx.x;
    if (i < n) p[i] = v;
}

// order-preserving float<->uint encoding for atomicMax-based segment max
__device__ __forceinline__ unsigned fenc(float f) {
    unsigned b = __float_as_uint(f);
    return (b & 0x80000000u) ? ~b : (b | 0x80000000u);
}
__device__ __forceinline__ float fdec(unsigned u) {
    return __uint_as_float((u & 0x80000000u) ? (u & 0x7FFFFFFFu) : ~u);
}

// edge fetch: first E0 edges from edge_index, then N self loops
__device__ __forceinline__ void edge_sd(const int* __restrict__ ei, int e,
                                        int& s, int& d) {
    if (e < E0) { s = ei[e]; d = ei[E0 + e]; }
    else        { s = e - E0; d = s; }
}

// ------------------------------------------------------------- WMMA GEMM
// C[M,Nc] = A[M,K] @ W[K,Nc] + bias   (f32 in/out, f16 WMMA, f32 accum)
// block = 256 thr = 8 waves -> 128x32 output tile; W tile staged transposed
// in LDS; each wave reuses one A fragment against two B fragments (2 WMMA).
__global__ __launch_bounds__(256) void gemm_wmma(
    const float* __restrict__ A, const float* __restrict__ W,
    const float* __restrict__ bias, float* __restrict__ C,
    int M, int K, int Nc)
{
    __shared__ _Float16 sB[32 * BSTR];        // [n][k] transposed, n<32, k<32
    const int tid  = threadIdx.x;
    const int lane = tid & 31;
    const int wave = tid >> 5;
    const int m0   = blockIdx.x * 128 + wave * 16;
    const int n0   = blockIdx.y * 32;
    const int rowA = m0 + (lane & 15);        // A row for this lane
    const int kSh  = (lane >> 4) * 8;         // K sub-group select (ISA 7.12.2)
    const int cn   = lane & 15;               // B/D column within 16-tile
    const bool fullN = (n0 + 32 <= Nc);       // block-uniform fast path

    v8f acc0 = {}, acc1 = {};
    for (int k0 = 0; k0 < K; k0 += 32) {
        // ---- stage W[k0:k0+32, n0:n0+32] transposed into LDS ----
        if (fullN) {
#pragma unroll
            for (int p = 0; p < 4; ++p) {
                int idx = p * 256 + tid;
                int kk = idx >> 5, nn = idx & 31;         // coalesced over nn
                sB[nn * BSTR + kk] =
                    (_Float16)W[(size_t)(k0 + kk) * Nc + n0 + nn];
            }
        } else {
#pragma unroll
            for (int p = 0; p < 4; ++p) {
                int idx = p * 256 + tid;
                int kk = idx >> 5, nn = idx & 31;
                int n  = n0 + nn;
                float w = (n < Nc) ? W[(size_t)(k0 + kk) * Nc + n] : 0.f;
                sB[nn * BSTR + kk] = (_Float16)w;
            }
        }
        __syncthreads();

        // ---- A fragment: two 16B + two 16B vector loads per lane ----
        v16h a;
        if (rowA < M) {
            const float* Ar = A + (size_t)rowA * K + k0 + kSh;
            float4 f0 = *(const float4*)(Ar);
            float4 f1 = *(const float4*)(Ar + 4);
            float4 f2 = *(const float4*)(Ar + 16);
            float4 f3 = *(const float4*)(Ar + 20);
            a[0]=(_Float16)f0.x;  a[1]=(_Float16)f0.y;
            a[2]=(_Float16)f0.z;  a[3]=(_Float16)f0.w;
            a[4]=(_Float16)f1.x;  a[5]=(_Float16)f1.y;
            a[6]=(_Float16)f1.z;  a[7]=(_Float16)f1.w;
            a[8]=(_Float16)f2.x;  a[9]=(_Float16)f2.y;
            a[10]=(_Float16)f2.z; a[11]=(_Float16)f2.w;
            a[12]=(_Float16)f3.x; a[13]=(_Float16)f3.y;
            a[14]=(_Float16)f3.z; a[15]=(_Float16)f3.w;
        } else {
#pragma unroll
            for (int i = 0; i < 16; ++i) a[i] = (_Float16)0.f;
        }

        // ---- B fragments: contiguous 16B LDS chunks (ds_load_b128) ----
        const _Float16* r0 = sB + cn * BSTR;         // columns n0+cn
        const _Float16* r1 = sB + (cn + 16) * BSTR;  // columns n0+16+cn
        v8h b0lo = *(const v8h*)(r0 + kSh);
        v8h b0hi = *(const v8h*)(r0 + kSh + 16);
        v8h b1lo = *(const v8h*)(r1 + kSh);
        v8h b1hi = *(const v8h*)(r1 + kSh + 16);
        v16h b0, b1;
#pragma unroll
        for (int i = 0; i < 8; ++i) {
            b0[i] = b0lo[i]; b0[8 + i] = b0hi[i];
            b1[i] = b1lo[i]; b1[8 + i] = b1hi[i];
        }
        acc0 = __builtin_amdgcn_wmma_f32_16x16x32_f16(
            false, a, false, b0, (short)0, acc0, false, false);
        acc1 = __builtin_amdgcn_wmma_f32_16x16x32_f16(
            false, a, false, b1, (short)0, acc1, false, false);
        __syncthreads();
    }

    // ---- store D + bias ----
    const int ncol0 = n0 + cn;
    const int ncol1 = n0 + 16 + cn;
    const float bv0 = (ncol0 < Nc) ? bias[ncol0] : 0.f;
    const float bv1 = (ncol1 < Nc) ? bias[ncol1] : 0.f;
    const int rb = m0 + ((lane >> 4) * 8);    // D: VGPR r -> M = rb + r
#pragma unroll
    for (int r = 0; r < 8; ++r) {
        int m = rb + r;
        if (m < M) {
            if (ncol0 < Nc) C[(size_t)m * Nc + ncol0] = acc0[r] + bv0;
            if (ncol1 < Nc) C[(size_t)m * Nc + ncol1] = acc1[r] + bv1;
        }
    }
}

// ------------------------------------------------------- edge attention
// e[edge,h] = sum_c leakyrelu(xl[src,h,c]+xr[dst,h,c]) * att[h,c]
// one wave32 per edge; lanes split into H groups; shfl_xor group reduce.
__global__ __launch_bounds__(256) void edge_attn(
    const int* __restrict__ ei, const float* __restrict__ xl,
    const float* __restrict__ xr, const float* __restrict__ att,
    float* __restrict__ e_out, unsigned* __restrict__ emax, int H, int C)
{
    int gw = (blockIdx.x * blockDim.x + threadIdx.x) >> 5;
    int lane = threadIdx.x & 31;
    if (gw >= ETOT) return;
    int s, d; edge_sd(ei, gw, s, d);
    int lph = 32 / H;                     // lanes per head (H in {1,4})
    int h = lane / lph, lh = lane % lph;
    const float* xls = xl + (size_t)s * H * C + h * C;
    const float* xrd = xr + (size_t)d * H * C + h * C;
    const float* at  = att + h * C;
    float sum = 0.f;
    int v4 = lph << 2;                    // floats covered per float4 sweep
    if ((C & (v4 - 1)) == 0) {            // vector path (layers 1,2)
        for (int c = lh * 4; c < C; c += v4) {
            float4 a4 = *(const float4*)(xls + c);
            float4 b4 = *(const float4*)(xrd + c);
            float4 t4 = *(const float4*)(at + c);
            float m0 = a4.x + b4.x, m1 = a4.y + b4.y;
            float m2 = a4.z + b4.z, m3 = a4.w + b4.w;
            m0 = (m0 >= 0.f) ? m0 : NEG_SLOPE * m0;
            m1 = (m1 >= 0.f) ? m1 : NEG_SLOPE * m1;
            m2 = (m2 >= 0.f) ? m2 : NEG_SLOPE * m2;
            m3 = (m3 >= 0.f) ? m3 : NEG_SLOPE * m3;
            sum += m0 * t4.x + m1 * t4.y + m2 * t4.z + m3 * t4.w;
        }
    } else {                              // scalar path (layer 3, C=10)
        for (int c = lh; c < C; c += lph) {
            float v = xls[c] + xrd[c];
            v = (v >= 0.f) ? v : NEG_SLOPE * v;
            sum += v * at[c];
        }
    }
    for (int off = lph >> 1; off > 0; off >>= 1)
        sum += __shfl_xor(sum, off, 32);
    if (lh == 0) {
        e_out[(size_t)gw * H + h] = sum;
        atomicMax(emax + (size_t)d * H + h, fenc(sum));
    }
}

// ee = exp(e - emax[dst]); denom[dst] += ee   (in-place over e buffer)
__global__ void edge_softmax(const int* __restrict__ ei,
                             float* __restrict__ e_io,
                             const unsigned* __restrict__ emax,
                             float* __restrict__ denom, int H)
{
    int i = blockIdx.x * blockDim.x + threadIdx.x;
    if (i >= ETOT * H) return;
    int eid = i / H, h = i - eid * H;
    int s, d; edge_sd(ei, eid, s, d); (void)s;
    float ex = __expf(e_io[i] - fdec(emax[(size_t)d * H + h]));
    e_io[i] = ex;
    atomicAdd(denom + (size_t)d * H + h, ex);
}

// acc[dst,h,c] += (ee/denom) * xl[src,h,c]; one wave32 per edge
__global__ __launch_bounds__(256) void edge_scatter(
    const int* __restrict__ ei, const float* __restrict__ xl,
    const float* __restrict__ ee, const float* __restrict__ denom,
    float* __restrict__ acc, int H, int C)
{
    int gw = (blockIdx.x * blockDim.x + threadIdx.x) >> 5;
    int lane = threadIdx.x & 31;
    if (gw >= ETOT) return;
    int s, d; edge_sd(ei, gw, s, d);
    const float* xs = xl + (size_t)s * H * C;
    float* ad = acc + (size_t)d * H * C;
    for (int h = 0; h < H; ++h) {
        float al = ee[(size_t)gw * H + h] / (denom[(size_t)d * H + h] + EPS_F);
        const float* xsh = xs + h * C;
        float*       adh = ad + h * C;
        for (int c = lane; c < C; c += 32)
            atomicAdd(adh + c, al * xsh[c]);
    }
}

// concat=False head-mean + bias + relu
__global__ void head_mean_bias_relu(const float* __restrict__ acc,
                                    const float* __restrict__ bias,
                                    float* __restrict__ out, int H, int C)
{
    int i = blockIdx.x * blockDim.x + threadIdx.x;
    if (i >= NNODES * C) return;
    int n = i / C, c = i - n * C;
    float s = 0.f;
    for (int h = 0; h < H; ++h) s += acc[(size_t)n * H * C + h * C + c];
    float v = s / (float)H + bias[c];
    out[i] = v > 0.f ? v : 0.f;
}

// in-place bias (+ optional relu) for concat=True / final layers
__global__ void bias_act(float* __restrict__ buf, const float* __restrict__ bias,
                         int count, int HC, int do_relu)
{
    int i = blockIdx.x * blockDim.x + threadIdx.x;
    if (i >= count) return;
    float v = buf[i] + bias[i % HC];
    buf[i] = (do_relu && v < 0.f) ? 0.f : v;
}

// ---------------------------------------------------------------- driver
extern "C" void kernel_launch(void* const* d_in, const int* in_sizes, int n_in,
                              void* d_out, int out_size, void* d_ws, size_t ws_size,
                              hipStream_t stream)
{
    (void)in_sizes; (void)n_in; (void)out_size; (void)ws_size;
    const float* x    = (const float*)d_in[0];
    const int*   ei   = (const int*)  d_in[1];
    const float* Wl1  = (const float*)d_in[2];
    const float* bl1  = (const float*)d_in[3];
    const float* Wr1  = (const float*)d_in[4];
    const float* br1  = (const float*)d_in[5];
    const float* att1 = (const float*)d_in[6];
    const float* b1   = (const float*)d_in[7];
    const float* Wl2  = (const float*)d_in[8];
    const float* bl2  = (const float*)d_in[9];
    const float* Wr2  = (const float*)d_in[10];
    const float* br2  = (const float*)d_in[11];
    const float* att2 = (const float*)d_in[12];
    const float* b2   = (const float*)d_in[13];
    const float* Wl3  = (const float*)d_in[14];
    const float* bl3  = (const float*)d_in[15];
    const float* Wr3  = (const float*)d_in[16];
    const float* br3  = (const float*)d_in[17];
    const float* att3 = (const float*)d_in[18];
    const float* b3   = (const float*)d_in[19];

    char* ws = (char*)d_ws;
    float*    xl   = (float*)(ws);                 // N*512 f32
    float*    xr   = (float*)(ws + 40960000);      // N*512 f32 (reused as L1 acc)
    float*    ebuf = (float*)(ws + 81920000);      // E*4 f32 (e -> ee in place)
    unsigned* emax = (unsigned*)(ws + 87360000);   // N*4 u32 (encoded max)
    float*    den  = (float*)(ws + 87680000);      // N*4 f32
    float*    h1   = (float*)(ws + 88000000);      // N*128 f32

    float* y_out = (float*)d_out;                  // [N,10]
    float* h_out = y_out + (size_t)NNODES * 10;    // [N,256]

    dim3 blk(256);
    const int edgeBlocks = (ETOT * 32 + 255) / 256;   // one wave per edge

    // ---------------- Layer 1: 512 -> (4,128), concat=False ----------------
    {
        const int K = 512, Nc = 512, H = 4, C = 128;
        dim3 g((NNODES + 127) / 128, (Nc + 31) / 32);
        gemm_wmma<<<g, blk, 0, stream>>>(x, Wl1, bl1, xl, NNODES, K, Nc);
        gemm_wmma<<<g, blk, 0, stream>>>(x, Wr1, br1, xr, NNODES, K, Nc);
        fill_u32<<<(NNODES*H + 255)/256, blk, 0, stream>>>(emax, 0x007FFFFFu, NNODES*H); // enc(-inf)
        zero_f32<<<(NNODES*H + 255)/256, blk, 0, stream>>>(den, NNODES*H);
        edge_attn<<<edgeBlocks, blk, 0, stream>>>(ei, xl, xr, att1, ebuf, emax, H, C);
        edge_softmax<<<(ETOT*H + 255)/256, blk, 0, stream>>>(ei, ebuf, emax, den, H);
        zero_f32<<<(NNODES*H*C + 255)/256, blk, 0, stream>>>(xr, NNODES*H*C); // xr dead -> acc
        edge_scatter<<<edgeBlocks, blk, 0, stream>>>(ei, xl, ebuf, den, xr, H, C);
        head_mean_bias_relu<<<(NNODES*C + 255)/256, blk, 0, stream>>>(xr, b1, h1, H, C);
    }
    // ---------------- Layer 2: 128 -> (4,64), concat=True ------------------
    {
        const int K = 128, Nc = 256, H = 4, C = 64;
        dim3 g((NNODES + 127) / 128, (Nc + 31) / 32);
        gemm_wmma<<<g, blk, 0, stream>>>(h1, Wl2, bl2, xl, NNODES, K, Nc);
        gemm_wmma<<<g, blk, 0, stream>>>(h1, Wr2, br2, xr, NNODES, K, Nc);
        fill_u32<<<(NNODES*H + 255)/256, blk, 0, stream>>>(emax, 0x007FFFFFu, NNODES*H);
        zero_f32<<<(NNODES*H + 255)/256, blk, 0, stream>>>(den, NNODES*H);
        edge_attn<<<edgeBlocks, blk, 0, stream>>>(ei, xl, xr, att2, ebuf, emax, H, C);
        edge_softmax<<<(ETOT*H + 255)/256, blk, 0, stream>>>(ei, ebuf, emax, den, H);
        zero_f32<<<(NNODES*H*C + 255)/256, blk, 0, stream>>>(h_out, NNODES*H*C);
        edge_scatter<<<edgeBlocks, blk, 0, stream>>>(ei, xl, ebuf, den, h_out, H, C);
        bias_act<<<(NNODES*H*C + 255)/256, blk, 0, stream>>>(h_out, b2, NNODES*H*C, H*C, 1);
    }
    // ---------------- Layer 3: 256 -> (1,10), concat=False -----------------
    {
        const int K = 256, Nc = 10, H = 1, C = 10;
        dim3 g((NNODES + 127) / 128, (Nc + 31) / 32);
        gemm_wmma<<<g, blk, 0, stream>>>(h_out, Wl3, bl3, xl, NNODES, K, Nc);
        gemm_wmma<<<g, blk, 0, stream>>>(h_out, Wr3, br3, xr, NNODES, K, Nc);
        fill_u32<<<(NNODES*H + 255)/256, blk, 0, stream>>>(emax, 0x007FFFFFu, NNODES*H);
        zero_f32<<<(NNODES*H + 255)/256, blk, 0, stream>>>(den, NNODES*H);
        edge_attn<<<edgeBlocks, blk, 0, stream>>>(ei, xl, xr, att3, ebuf, emax, H, C);
        edge_softmax<<<(ETOT*H + 255)/256, blk, 0, stream>>>(ei, ebuf, emax, den, H);
        zero_f32<<<(NNODES*H*C + 255)/256, blk, 0, stream>>>(y_out, NNODES*H*C);
        edge_scatter<<<edgeBlocks, blk, 0, stream>>>(ei, xl, ebuf, den, y_out, H, C);
        bias_act<<<(NNODES*H*C + 255)/256, blk, 0, stream>>>(y_out, b3, NNODES*H*C, H*C, 0);
    }
}
